// Mamba_32049045963363
// MI455X (gfx1250) — compile-verified, matrix operand order checked by reading, not run on hardware
//
#include <hip/hip_runtime.h>
#include <hip/hip_bf16.h>
#include <stddef.h>

// ---------------------------------------------------------------------------
// Mamba2 forward (B=1, L=1024, D_MODEL=768, 4 layers) for gfx1250 / MI455X.
// GEMMs: v_wmma_f32_16x16x32_bf16, 64x256 block tile, 32x64 wave tile
// (8 WMMAs / wave / K-step), double-buffered LDS staging (fp32->bf16 on the
// global->LDS path), one barrier per K iteration.
// Scan: sequential recurrence, one workgroup per head, 64x128 state in
// registers; per-step B/C/x rows staged via GLOBAL_LOAD_ASYNC_TO_LDS
// (ASYNCcnt + s_wait_asynccnt) when the toolchain exposes the builtin.
// ---------------------------------------------------------------------------

#define D_MODEL   768
#define L_SEQ     1024
#define N_LAYERS  4
#define VOCAB     32000
#define D_INNER   1536
#define HEADDIM   64
#define NHEADS    24
#define D_STATE   128
#define DCONV     4
#define CONV_DIM  (D_INNER + 2 * D_STATE)              // 1792
#define D_IN_PROJ (2 * D_INNER + 2 * D_STATE + NHEADS) // 3352
#define EPSF      1e-5f

typedef __bf16 bf16;
typedef bf16  bf16x4  __attribute__((ext_vector_type(4)));
typedef bf16  bf16x8  __attribute__((ext_vector_type(8)));
typedef bf16  bf16x16 __attribute__((ext_vector_type(16)));
typedef float f32x4   __attribute__((ext_vector_type(4)));
typedef float v8f     __attribute__((ext_vector_type(8)));

#if defined(__has_builtin)
#if __has_builtin(__builtin_amdgcn_global_load_async_to_lds_b32) && \
    __has_builtin(__builtin_amdgcn_s_wait_asynccnt)
#define USE_ASYNC_LDS 1
#endif
#endif
#ifndef USE_ASYNC_LDS
#define USE_ASYNC_LDS 0
#endif

#if USE_ASYNC_LDS
typedef __attribute__((address_space(1))) int as1_i32;
typedef __attribute__((address_space(3))) int as3_i32;
#endif

// ------------------------------ WMMA GEMM ----------------------------------
// C[M,N] = A[M,K] * W[N,K]^T  (A, W, C fp32 row-major; K % 32 == 0,
// M % 64 == 0; N guarded).
#define BM 64
#define BN 256
#define BK 32
#define LDT 40 // LDS row stride in bf16 elements (80B: 16B aligned, conflict-free)

__global__ __launch_bounds__(256)
void gemm_xw_bf16(const float* __restrict__ A, const float* __restrict__ W,
                  float* __restrict__ C, int M, int N, int K) {
  __shared__ __align__(16) bf16 As[2][BM * LDT];
  __shared__ __align__(16) bf16 Ws[2][BN * LDT];

  const int tid  = threadIdx.x;
  const int bm   = blockIdx.y * BM;
  const int bn   = blockIdx.x * BN;
  const int wave = tid >> 5;
  const int lane = tid & 31;
  const int half = lane >> 4;   // 0: lanes 0-15, 1: lanes 16-31
  const int l16  = lane & 15;
  const int wm   = (wave & 1) * 32;   // 2 waves in M
  const int wn   = (wave >> 1) * 64;  // 4 waves in N, 64 cols each

  v8f acc[2][4] = {}; // fp32 accumulators, ISA 16x16 f32 C/D layout

  // stage tile at K-offset k0 into LDS buffer `buf` (fp32 -> bf16)
  auto stage = [&](int buf, int k0) {
#pragma unroll
    for (int it = 0; it < 2; ++it) {               // A: 64 rows x 8 float4
      int idx = tid * 2 + it;                      // 0..511
      int r = idx >> 3, c4 = idx & 7;
      f32x4 v = *(const f32x4*)(A + (size_t)(bm + r) * K + k0 + c4 * 4);
      bf16x4 b = { (bf16)v[0], (bf16)v[1], (bf16)v[2], (bf16)v[3] };
      *(bf16x4*)(&As[buf][r * LDT + c4 * 4]) = b;
    }
#pragma unroll
    for (int it = 0; it < 8; ++it) {               // W: 256 rows x 8 float4
      int idx = tid * 8 + it;                      // 0..2047
      int r = idx >> 3, c4 = idx & 7;
      int gr = bn + r;
      f32x4 v = {0.f, 0.f, 0.f, 0.f};
      if (gr < N) v = *(const f32x4*)(W + (size_t)gr * K + k0 + c4 * 4);
      bf16x4 b = { (bf16)v[0], (bf16)v[1], (bf16)v[2], (bf16)v[3] };
      *(bf16x4*)(&Ws[buf][r * LDT + c4 * 4]) = b;
    }
  };

  const int nk = K / BK;
  stage(0, 0);
  __syncthreads();

  for (int kt = 0; kt < nk; ++kt) {
    const int cur = kt & 1;
    if (kt + 1 < nk) {
      stage(cur ^ 1, (kt + 1) * BK);               // overlap with WMMAs below
      if (kt + 2 < nk) {                           // speculative L2 prefetch
        __builtin_prefetch(A + (size_t)(bm + (tid & 63)) * K + (kt + 2) * BK, 0, 0);
        if (bn + tid < N)
          __builtin_prefetch(W + (size_t)(bn + tid) * K + (kt + 2) * BK, 0, 0);
      }
    }

    // fragments per ISA 7.12.2 (16-bit A 16x32 / B 32x16 layouts):
    // lane half h, m/n = lane&15; elems 0..7 -> K = h*8 + e,
    // elems 8..15 -> K = 16 + h*8 + (e-8). Two 16B LDS reads per fragment.
    union Frag { bf16x16 v; bf16x8 h[2]; };
    Frag fa[2], fb[4];
#pragma unroll
    for (int i = 0; i < 2; ++i) {
      const bf16* base = &As[cur][(wm + i * 16 + l16) * LDT];
      fa[i].h[0] = *(const bf16x8*)(base + half * 8);
      fa[i].h[1] = *(const bf16x8*)(base + 16 + half * 8);
    }
#pragma unroll
    for (int j = 0; j < 4; ++j) {
      const bf16* base = &Ws[cur][(wn + j * 16 + l16) * LDT];
      fb[j].h[0] = *(const bf16x8*)(base + half * 8);
      fb[j].h[1] = *(const bf16x8*)(base + 16 + half * 8);
    }
#pragma unroll
    for (int i = 0; i < 2; ++i)
#pragma unroll
      for (int j = 0; j < 4; ++j)
        acc[i][j] = __builtin_amdgcn_wmma_f32_16x16x32_bf16(
            false, fa[i].v, false, fb[j].v, (short)0, acc[i][j], false, false);
    __syncthreads();
  }

  // store: C/D layout -> VGPR r holds M = r + half*8, N = l16
#pragma unroll
  for (int i = 0; i < 2; ++i)
#pragma unroll
    for (int j = 0; j < 4; ++j) {
      int col = bn + wn + j * 16 + l16;
      if (col < N) {
#pragma unroll
        for (int r = 0; r < 8; ++r) {
          int row = bm + wm + i * 16 + half * 8 + r;
          C[(size_t)row * N + col] = acc[i][j][r];
        }
      }
    }
}

// ------------------------------ elementwise --------------------------------

__global__ void zero_kernel(float* p, int n) {
  int i = blockIdx.x * 256 + threadIdx.x;
  if (i < n) p[i] = 0.f;
}

__global__ void embed_kernel(const int* __restrict__ ids,
                             const float* __restrict__ ew,
                             float* __restrict__ out) {
  int t = blockIdx.x;
  int id = ids[t];
#pragma unroll
  for (int i = threadIdx.x; i < D_MODEL; i += 256)
    out[(size_t)t * D_MODEL + i] = ew[(size_t)id * D_MODEL + i];
}

// residual += hidden; out = rmsnorm(residual) * w   (row of 768)
__global__ __launch_bounds__(256)
void add_rmsnorm_kernel(float* __restrict__ residual,
                        const float* __restrict__ hidden,
                        const float* __restrict__ w,
                        float* __restrict__ out) {
  int t = blockIdx.x;
  __shared__ float red[8];
  float v[3];
  float ss = 0.f;
#pragma unroll
  for (int i = 0; i < 3; ++i) {
    int d = threadIdx.x + i * 256;
    float x = residual[(size_t)t * D_MODEL + d] + hidden[(size_t)t * D_MODEL + d];
    residual[(size_t)t * D_MODEL + d] = x;
    v[i] = x;
    ss += x * x;
  }
#pragma unroll
  for (int off = 16; off > 0; off >>= 1) ss += __shfl_down(ss, off, 32);
  if ((threadIdx.x & 31) == 0) red[threadIdx.x >> 5] = ss;
  __syncthreads();
  float s = 0.f;
#pragma unroll
  for (int i = 0; i < 8; ++i) s += red[i];
  float scale = rsqrtf(s / (float)D_MODEL + EPSF);
#pragma unroll
  for (int i = 0; i < 3; ++i) {
    int d = threadIdx.x + i * 256;
    out[(size_t)t * D_MODEL + d] = v[i] * scale * w[d];
  }
}

// dt = softplus(zx[:, -24:] + dt_bias); dA = exp(dt * (-exp(A_log)))
__global__ void dt_kernel(const float* __restrict__ zx,
                          const float* __restrict__ dtb,
                          const float* __restrict__ alog,
                          float* __restrict__ dt, float* __restrict__ dA) {
  int i = blockIdx.x * 256 + threadIdx.x;
  if (i >= L_SEQ * NHEADS) return;
  int t = i / NHEADS, h = i % NHEADS;
  float raw = zx[(size_t)t * D_IN_PROJ + (D_IN_PROJ - NHEADS) + h] + dtb[h];
  float sp = (raw > 20.f) ? raw : log1pf(expf(raw));
  float A = -expf(alog[h]);
  dt[i] = sp;
  dA[i] = expf(sp * A);
}

// causal depthwise conv (DCONV=4) over zx[:, 1536:3328] + bias, then silu
__global__ __launch_bounds__(256)
void conv_silu_kernel(const float* __restrict__ zx,
                      const float* __restrict__ cw,
                      const float* __restrict__ cb,
                      float* __restrict__ out) {
  int c = blockIdx.x * 256 + threadIdx.x;
  int t = blockIdx.y;
  if (c >= CONV_DIM) return;
  float acc = cb[c];
#pragma unroll
  for (int k = 0; k < DCONV; ++k) {
    int tt = t + k - (DCONV - 1);
    float x = (tt >= 0) ? zx[(size_t)tt * D_IN_PROJ + D_INNER + c] : 0.f;
    acc += x * cw[c * DCONV + k];
  }
  out[(size_t)t * CONV_DIM + c] = acc / (1.f + expf(-acc));
}

// sequential SSM scan: one block per head. state[64][128] lives in registers
// (32 f32/thread). thread -> (p = tid/4, n-quarter = tid%4).
__global__ __launch_bounds__(256)
void scan_kernel(const float* __restrict__ xbc,
                 const float* __restrict__ dt,
                 const float* __restrict__ dA,
                 const float* __restrict__ Dp,
                 float* __restrict__ y) {
  int h = blockIdx.x;
  int tid = threadIdx.x;
  int p = tid >> 2, nq = tid & 3;
  __shared__ float sBC[2 * D_STATE];  // [0,128): B, [128,256): C
  __shared__ float sX[HEADDIM], partial[256];
  float st[32];
#pragma unroll
  for (int j = 0; j < 32; ++j) st[j] = 0.f;
  float Dh = Dp[h];

  for (int t = 0; t < L_SEQ; ++t) {
    const float* row = xbc + (size_t)t * CONV_DIM;
#if USE_ASYNC_LDS
    // gfx1250 async copy global->LDS (ASYNCcnt); B and C are contiguous.
    if (tid < 2 * D_STATE) {
      __builtin_amdgcn_global_load_async_to_lds_b32(
          (as1_i32*)(row + D_INNER + tid),
          (as3_i32*)(&sBC[tid]), 0, 0);
    } else if (tid < 2 * D_STATE + HEADDIM) {
      int i = tid - 2 * D_STATE;
      __builtin_amdgcn_global_load_async_to_lds_b32(
          (as1_i32*)(row + h * HEADDIM + i),
          (as3_i32*)(&sX[i]), 0, 0);
    }
    __builtin_amdgcn_s_wait_asynccnt(0);
#else
    if (tid < 2 * D_STATE) sBC[tid] = row[D_INNER + tid];
    if (tid < HEADDIM)     sX[tid]  = row[h * HEADDIM + tid];
#endif
    __syncthreads();

    float dAt = dA[t * NHEADS + h];
    float dtt = dt[t * NHEADS + h];
    float xp  = sX[p];
    float dx  = dtt * xp;
    float part = 0.f;
#pragma unroll
    for (int j = 0; j < 32; ++j) {
      int n = nq * 32 + j;
      float s = dAt * st[j] + dx * sBC[n];
      st[j] = s;
      part += s * sBC[D_STATE + n];
    }
    partial[tid] = part;
    __syncthreads();
    if (nq == 0)
      y[(size_t)t * D_INNER + h * HEADDIM + p] =
          partial[tid] + partial[tid + 1] + partial[tid + 2] + partial[tid + 3] +
          Dh * xp;
  }
}

// out = rmsnorm(y * silu(z)) * gw   (row of 1536; z = zx[:, :1536])
__global__ __launch_bounds__(256)
void gated_rmsnorm_kernel(const float* __restrict__ y,
                          const float* __restrict__ zx,
                          const float* __restrict__ gw,
                          float* __restrict__ out) {
  int t = blockIdx.x;
  __shared__ float red[8];
  float v[6];
  float ss = 0.f;
#pragma unroll
  for (int i = 0; i < 6; ++i) {
    int d = threadIdx.x + i * 256;
    float z = zx[(size_t)t * D_IN_PROJ + d];
    float u = y[(size_t)t * D_INNER + d] * (z / (1.f + expf(-z)));
    v[i] = u;
    ss += u * u;
  }
#pragma unroll
  for (int off = 16; off > 0; off >>= 1) ss += __shfl_down(ss, off, 32);
  if ((threadIdx.x & 31) == 0) red[threadIdx.x >> 5] = ss;
  __syncthreads();
  float s = 0.f;
#pragma unroll
  for (int i = 0; i < 8; ++i) s += red[i];
  float scale = rsqrtf(s / (float)D_INNER + EPSF);
#pragma unroll
  for (int i = 0; i < 6; ++i) {
    int d = threadIdx.x + i * 256;
    out[(size_t)t * D_INNER + d] = v[i] * scale * gw[d];
  }
}

// ------------------------------- driver ------------------------------------

extern "C" void kernel_launch(void* const* d_in, const int* in_sizes, int n_in,
                              void* d_out, int out_size, void* d_ws, size_t ws_size,
                              hipStream_t stream) {
  const int*   ids        = (const int*)d_in[0];
  const float* embed_w    = (const float*)d_in[1];
  const float* norm_w     = (const float*)d_in[2];
  const float* in_proj_w  = (const float*)d_in[3];
  const float* conv_w     = (const float*)d_in[4];
  const float* conv_b     = (const float*)d_in[5];
  const float* dt_bias    = (const float*)d_in[6];
  const float* A_log      = (const float*)d_in[7];
  const float* D_param    = (const float*)d_in[8];
  const float* gnorm_w    = (const float*)d_in[9];
  const float* out_proj_w = (const float*)d_in[10];
  const float* norm_f_w   = (const float*)d_in[11];
  const float* lm_head_w  = (const float*)d_in[12];
  float* out = (float*)d_out;

  float* ws = (float*)d_ws;
  float* residual = ws; ws += (size_t)L_SEQ * D_MODEL;
  float* hidden   = ws; ws += (size_t)L_SEQ * D_MODEL;
  float* hnorm    = ws; ws += (size_t)L_SEQ * D_MODEL;
  float* zx       = ws; ws += (size_t)L_SEQ * D_IN_PROJ;
  float* xconv    = ws; ws += (size_t)L_SEQ * CONV_DIM;
  float* dtv      = ws; ws += (size_t)L_SEQ * NHEADS;
  float* dAv      = ws; ws += (size_t)L_SEQ * NHEADS;
  float* yv       = ws; ws += (size_t)L_SEQ * D_INNER;
  float* ynorm    = ws; ws += (size_t)L_SEQ * D_INNER;

  zero_kernel<<<(L_SEQ * D_MODEL + 255) / 256, 256, 0, stream>>>(residual, L_SEQ * D_MODEL);
  embed_kernel<<<L_SEQ, 256, 0, stream>>>(ids, embed_w, hidden);

  for (int i = 0; i < N_LAYERS; ++i) {
    add_rmsnorm_kernel<<<L_SEQ, 256, 0, stream>>>(residual, hidden,
                                                  norm_w + i * D_MODEL, hnorm);
    // in_proj: (1024x768) x (3352x768)^T
    gemm_xw_bf16<<<dim3((D_IN_PROJ + BN - 1) / BN, L_SEQ / BM), 256, 0, stream>>>(
        hnorm, in_proj_w + (size_t)i * D_IN_PROJ * D_MODEL, zx,
        L_SEQ, D_IN_PROJ, D_MODEL);
    dt_kernel<<<(L_SEQ * NHEADS + 255) / 256, 256, 0, stream>>>(
        zx, dt_bias + i * NHEADS, A_log + i * NHEADS, dtv, dAv);
    conv_silu_kernel<<<dim3((CONV_DIM + 255) / 256, L_SEQ), 256, 0, stream>>>(
        zx, conv_w + (size_t)i * CONV_DIM * DCONV, conv_b + (size_t)i * CONV_DIM,
        xconv);
    scan_kernel<<<NHEADS, 256, 0, stream>>>(xconv, dtv, dAv,
                                            D_param + i * NHEADS, yv);
    gated_rmsnorm_kernel<<<L_SEQ, 256, 0, stream>>>(yv, zx,
                                                    gnorm_w + i * D_INNER, ynorm);
    // out_proj: (1024x1536) x (768x1536)^T
    gemm_xw_bf16<<<dim3((D_MODEL + BN - 1) / BN, L_SEQ / BM), 256, 0, stream>>>(
        ynorm, out_proj_w + (size_t)i * D_MODEL * D_INNER, hidden,
        L_SEQ, D_MODEL, D_INNER);
  }

  add_rmsnorm_kernel<<<L_SEQ, 256, 0, stream>>>(residual, hidden, norm_f_w, hnorm);
  // lm_head: (1024x768) x (32000x768)^T
  gemm_xw_bf16<<<dim3(VOCAB / BN, L_SEQ / BM), 256, 0, stream>>>(
      hnorm, lm_head_w, out, L_SEQ, VOCAB, D_MODEL);
}
